// Transolver_block_69002944578054
// MI455X (gfx1250) — matrix operand order repaired and checked by reading, not compiled, not run on hardware
//
#include <hip/hip_runtime.h>
#include <math.h>

// ---------------------------------------------------------------------------
// Types for CDNA5 WMMA (wave32)
// ---------------------------------------------------------------------------
typedef __attribute__((ext_vector_type(16))) __bf16   v16bf;
typedef __attribute__((ext_vector_type(8)))  float    v8f;
typedef __attribute__((ext_vector_type(8)))  unsigned v8u;

#define LOGNEG 2.6256905f   // log(-log(1e-6))

// gfx1250 async global->LDS path (guarded: fall back to sync staging)
#if defined(__has_builtin)
#  if __has_builtin(__builtin_amdgcn_global_load_async_to_lds_b128) && \
      __has_builtin(__builtin_amdgcn_s_wait_asynccnt)
#    define HAVE_ASYNC_LDS 1
#  endif
#endif
#ifndef HAVE_ASYNC_LDS
#  define HAVE_ASYNC_LDS 0
#endif

#if HAVE_ASYNC_LDS
// builtin signature (from clang diagnostic): (v4i __device__*, v4i __shared__*, imm int, imm int)
typedef int v4i_t __attribute__((vector_size(4 * sizeof(int))));
typedef __attribute__((address_space(1))) v4i_t gv4i_t;   // global (prints as __device__)
typedef __attribute__((address_space(3))) v4i_t lv4i_t;   // LDS    (prints as __shared__)
#endif

static __device__ __forceinline__ float gelu_exact(float x) {
    return 0.5f * x * (1.0f + erff(x * 0.70710678118f));
}

// ---------------------------------------------------------------------------
// Weight prep: f32 [K,N] -> bf16 [N,K] (transposed so WMMA B loads are dwords)
// ---------------------------------------------------------------------------
__global__ void k_wT(const float* __restrict__ W, __bf16* __restrict__ Wt, int K, int N) {
    int t = blockIdx.x * 256 + threadIdx.x;
    if (t >= K * N) return;
    int k = t / N, n = t % N;
    Wt[(size_t)n * K + k] = (__bf16)W[t];
}

// ---------------------------------------------------------------------------
// LayerNorm over C=512, one wave per row, bf16 output
// ---------------------------------------------------------------------------
__global__ __launch_bounds__(256) void k_ln_big(const float* __restrict__ x,
                                                const float* __restrict__ g,
                                                const float* __restrict__ b,
                                                __bf16* __restrict__ out, int rows) {
    int wave = threadIdx.x >> 5, lane = threadIdx.x & 31;
    int row = blockIdx.x * 8 + wave;
    if (row >= rows) return;
    const float* xr = x + (size_t)row * 512;
    float v[16], s = 0.f, s2 = 0.f;
#pragma unroll
    for (int i = 0; i < 16; ++i) { v[i] = xr[lane + 32 * i]; s += v[i]; s2 += v[i] * v[i]; }
#pragma unroll
    for (int off = 16; off > 0; off >>= 1) {
        s  += __shfl_xor(s,  off, 32);
        s2 += __shfl_xor(s2, off, 32);
    }
    float m  = s  * (1.f / 512.f);
    float var = s2 * (1.f / 512.f) - m * m;
    float rs = rsqrtf(var + 1e-5f);
#pragma unroll
    for (int i = 0; i < 16; ++i) {
        int idx = lane + 32 * i;
        out[(size_t)row * 512 + idx] = (__bf16)((v[i] - m) * rs * g[idx] + b[idx]);
    }
}

// ---------------------------------------------------------------------------
// Generic bf16 WMMA GEMM: out[M,N] = A[M,K] @ Bt[N,K]^T (+bias) with epilogues
// Block = 256 thr = 8 M-waves -> tile 128(M) x 64(N).
// B tile (64x32 bf16 per K-step) is staged into LDS, double-buffered via
// GLOBAL_LOAD_ASYNC_TO_LDS_B128 (ASYNCcnt) when available.
// Each wave: one A fragment -> 4 x v_wmma_f32_16x16x32_bf16 per K-step.
// ---------------------------------------------------------------------------
enum { EPI_F32 = 0, EPI_BF16 = 1, EPI_XP = 2, EPI_GELU_BF16 = 3, EPI_RESID_F32 = 4, EPI_LOGIT_T = 5 };

template<int EPI>
__global__ __launch_bounds__(256)
void k_wmma_gemm(const __bf16* __restrict__ A, const __bf16* __restrict__ Bt,
                 const float* __restrict__ bias,
                 float* __restrict__ outf, __bf16* __restrict__ outh,
                 const float* __restrict__ extra,   // residual (f32[M*N]) or colscale for LOGIT_T
                 int M, int K, int N, int Ntok, int Hh) {
    __shared__ __attribute__((aligned(16))) __bf16 ldsB[2][64][32];

    const int t = threadIdx.x;
    const int lane = t & 31, wave = t >> 5;
    const int tilesN = N >> 6;
    const int mB = (blockIdx.x / tilesN) * 128 + wave * 16;
    const int nB = (blockIdx.x % tilesN) * 64;
    const int half = lane >> 4, l16 = lane & 15;
    const int am = mB + l16;
    const unsigned* pa = (const unsigned*)(A + (size_t)am * K);

    // B staging: thread t copies 16B: row sn (0..63), bf16 chunk sc
    const int sn = t >> 2, sc = (t & 3) * 8;
    const __bf16* gB = Bt + (size_t)(nB + sn) * K + sc;

    v8f acc[4] = {};

    auto stage = [&](int buf, int kk) {
#if HAVE_ASYNC_LDS
        __builtin_amdgcn_global_load_async_to_lds_b128(
            (gv4i_t*)(gB + kk),
            (lv4i_t*)&ldsB[buf][sn][sc], 0, 0);
#else
        uint4 v = *(const uint4*)(gB + kk);
        *(uint4*)&ldsB[buf][sn][sc] = v;
#endif
    };

    stage(0, 0);
    int buf = 0;
    for (int kk = 0; kk < K; kk += 32) {
        if (kk + 32 < K) stage(buf ^ 1, kk + 32);
#if HAVE_ASYNC_LDS
        if (kk + 32 < K) __builtin_amdgcn_s_wait_asynccnt(1);
        else             __builtin_amdgcn_s_wait_asynccnt(0);
#endif
        __syncthreads();

        v8u au;
#pragma unroll
        for (int v = 0; v < 4; ++v) {                   // ISA A 16x32 bf16 layout
            int ka = kk + half * 8 + 2 * v;
            au[v]     = pa[ka >> 1];
            au[v + 4] = pa[(ka + 16) >> 1];
        }
        v16bf a = __builtin_bit_cast(v16bf, au);
        if (kk + 32 < K)                                // hide HBM latency of next A slice
            __builtin_prefetch((const void*)(pa + ((kk + 32) >> 1)), 0, 3);

#pragma unroll
        for (int j = 0; j < 4; ++j) {                   // ISA B 32x16 bf16 layout (from LDS)
            const unsigned* pb = (const unsigned*)&ldsB[buf][j * 16 + l16][0];
            v8u bu;
#pragma unroll
            for (int v = 0; v < 8; ++v) bu[v] = pb[(v >> 2) * 8 + half * 4 + (v & 3)];
            v16bf b = __builtin_bit_cast(v16bf, bu);
            acc[j] = __builtin_amdgcn_wmma_f32_16x16x32_bf16(false, a, false, b, (short)0, acc[j], false, false);
        }
        __syncthreads();
        buf ^= 1;
    }

#pragma unroll
    for (int j = 0; j < 4; ++j) {
        int n = nB + j * 16 + l16;
        float bb = bias ? bias[n] : 0.f;
#pragma unroll
        for (int r = 0; r < 8; ++r) {                   // ISA C/D 16x16 f32 layout
            int m = mB + half * 8 + r;
            float v0 = acc[j][r] + bb;
            if constexpr (EPI == EPI_F32) {
                outf[(size_t)m * N + n] = v0;
            } else if constexpr (EPI == EPI_BF16) {
                outh[(size_t)m * N + n] = (__bf16)v0;
            } else if constexpr (EPI == EPI_XP) {
                outf[(size_t)m * N + n] = v0;
                outh[(size_t)m * N + n] = (__bf16)v0;
            } else if constexpr (EPI == EPI_GELU_BF16) {
                outh[(size_t)m * N + n] = (__bf16)gelu_exact(v0);
            } else if constexpr (EPI == EPI_RESID_F32) {
                outf[(size_t)m * N + n] = v0 + extra[(size_t)m * N + n];
            } else if constexpr (EPI == EPI_LOGIT_T) {
                // row m = (b*Ntok + n_tok)*Hh + h ; write sw^T[b,h, g, n_tok]
                int b   = m / (Ntok * Hh);
                int rem = m - b * Ntok * Hh;
                int ntk = rem / Hh;
                int h   = rem - ntk * Hh;
                float sc2 = extra[((size_t)(b * Hh + h)) * Ntok + ntk];
                size_t base = ((size_t)(b * Hh + h) * N) * Ntok + ntk;
                outf[base + (size_t)n * Ntok] = (v0 - LOGNEG) * sc2;
            }
        }
    }
}

// ---------------------------------------------------------------------------
// Ada-Temp: invt[(b,h,n)] = 1/(0.5 + clip(xp_row . w_ada + b_ada, -.4, .4))
// wave-per-row, coalesced
// ---------------------------------------------------------------------------
__global__ __launch_bounds__(256) void k_temp(const float* __restrict__ xp,
                                              const float* __restrict__ wada,
                                              const float* __restrict__ bada,
                                              float* __restrict__ invt,
                                              int Ntok, int Hh, int total) {
    int wave = threadIdx.x >> 5, lane = threadIdx.x & 31;
    int t = blockIdx.x * 8 + wave;
    if (t >= total) return;
    int b = t / (Hh * Ntok);
    int rem = t - b * Hh * Ntok;
    int h = rem / Ntok;
    int n = rem - h * Ntok;
    const float* r = xp + (((size_t)b * Ntok + n) * Hh + h) * 64;
    float s = r[lane] * wada[lane] + r[lane + 32] * wada[lane + 32];
#pragma unroll
    for (int off = 16; off > 0; off >>= 1) s += __shfl_xor(s, off, 32);
    if (lane == 0) {
        s = fminf(0.4f, fmaxf(-0.4f, s + bada[0]));
        invt[t] = 1.f / (0.5f + s);
    }
}

// ---------------------------------------------------------------------------
// Softmax over contiguous rows of length Ntok (sw^T layout [B,H,G,N])
// ---------------------------------------------------------------------------
__global__ __launch_bounds__(256) void k_softmax(float* __restrict__ swT, int Ntok) {
    float* row = swT + (size_t)blockIdx.x * Ntok;
    __shared__ float red[256];
    int t = threadIdx.x;
    float mx = -1e30f;
    for (int i = t; i < Ntok; i += 256) mx = fmaxf(mx, row[i]);
    red[t] = mx; __syncthreads();
    for (int s = 128; s > 0; s >>= 1) { if (t < s) red[t] = fmaxf(red[t], red[t + s]); __syncthreads(); }
    mx = red[0]; __syncthreads();
    float sum = 0.f;
    for (int i = t; i < Ntok; i += 256) sum += expf(row[i] - mx);
    red[t] = sum; __syncthreads();
    for (int s = 128; s > 0; s >>= 1) { if (t < s) red[t] += red[t + s]; __syncthreads(); }
    float inv = 1.f / red[0];
    for (int i = t; i < Ntok; i += 256) row[i] = expf(row[i] - mx) * inv;
}

// ---------------------------------------------------------------------------
// Slice tokens: es[(b,h,g),c] = sum_n xp * sw / (snorm + 1e-5)
// Split over N-chunks with f32 atomics for parallelism, then finalize divide.
// ---------------------------------------------------------------------------
__global__ void k_zero(float* __restrict__ p, int n) {
    int t = blockIdx.x * 256 + threadIdx.x;
    if (t < n) p[t] = 0.f;
}

__global__ __launch_bounds__(256) void k_es_partial(const float* __restrict__ xp,
                                                    const float* __restrict__ swT,
                                                    float* __restrict__ esA,
                                                    float* __restrict__ snorm,
                                                    int Ntok, int Hh, int nch) {
    int bh = blockIdx.y;
    int b = bh >> 3, h = bh & 7;
    int gbase = blockIdx.x * 16;
    int chunk = Ntok / nch;
    int n0 = blockIdx.z * chunk, n1 = n0 + chunk;
    int t = threadIdx.x, c = t & 63, gs = t >> 6;
    float acc[4] = {0, 0, 0, 0}, sn[4] = {0, 0, 0, 0};
    const float* xpb = xp + ((size_t)b * Ntok * Hh + h) * 64;
    const float* swb = swT + (size_t)bh * 64 * Ntok;
    for (int n = n0; n < n1; ++n) {
        float xv = xpb[(size_t)n * Hh * 64 + c];
#pragma unroll
        for (int j = 0; j < 4; ++j) {
            int g = gbase + gs * 4 + j;
            float sv = swb[(size_t)g * Ntok + n];
            acc[j] += xv * sv;
            if (c == 0) sn[j] += sv;
        }
    }
#pragma unroll
    for (int j = 0; j < 4; ++j) {
        int g = gbase + gs * 4 + j;
        atomicAdd(&esA[((size_t)bh * 64 + g) * 64 + c], acc[j]);
        if (c == 0) atomicAdd(&snorm[bh * 64 + g], sn[j]);
    }
}

__global__ void k_es_div(float* __restrict__ es, const float* __restrict__ snorm) {
    int t = blockIdx.x * 256 + threadIdx.x;
    if (t >= 16 * 64 * 64) return;
    es[t] /= (snorm[t >> 6] + 1e-5f);
}

// ---------------------------------------------------------------------------
// Erwin helpers (tiny tensors, fp32 VALU path)
// ---------------------------------------------------------------------------
__global__ void k_rank(const float* __restrict__ es, int* __restrict__ rank) {
    int bh = blockIdx.x, i = threadIdx.x;
    __shared__ float key[64];
    float ki = es[((size_t)bh * 64 + i) * 64 + 0];
    key[i] = ki; __syncthreads();
    int r = 0;
    for (int j = 0; j < 64; ++j) {
        float kj = key[j];
        r += (kj < ki) || (kj == ki && j < i);   // stable argsort rank
    }
    rank[bh * 64 + i] = r;
}

__global__ void k_gather(const float* __restrict__ es, const int* __restrict__ rank,
                         float* __restrict__ xg) {
    int t = blockIdx.x * 256 + threadIdx.x;
    if (t >= 16 * 64 * 64) return;
    int bh = t >> 12, i = (t >> 6) & 63, c = t & 63;
    xg[((size_t)bh * 64 + rank[bh * 64 + i]) * 64 + c] = es[t];
}

__global__ void k_proc_t(const float* __restrict__ procg, const int* __restrict__ rank,
                         __bf16* __restrict__ procT) {
    int t = blockIdx.x * 256 + threadIdx.x;
    if (t >= 16 * 64 * 64) return;
    int bh = t >> 12, i = (t >> 6) & 63, c = t & 63;
    procT[(size_t)bh * 4096 + c * 64 + i] =
        (__bf16)procg[((size_t)bh * 64 + rank[bh * 64 + i]) * 64 + c];
}

__global__ void k_ln_small(const float* __restrict__ x, const float* __restrict__ g,
                           const float* __restrict__ b, float* __restrict__ out, int c) {
    int row = blockIdx.x, t = threadIdx.x;
    __shared__ float red[128];
    float v = x[(size_t)row * c + t];
    red[t] = v; __syncthreads();
    for (int s = c >> 1; s > 0; s >>= 1) { if (t < s) red[t] += red[t + s]; __syncthreads(); }
    float m = red[0] / c; __syncthreads();
    float d = (v - m) * (v - m);
    red[t] = d; __syncthreads();
    for (int s = c >> 1; s > 0; s >>= 1) { if (t < s) red[t] += red[t + s]; __syncthreads(); }
    float var = red[0] / c;
    out[(size_t)row * c + t] = (v - m) * rsqrtf(var + 1e-5f) * g[t] + b[t];
}

__global__ void k_sgemm(const float* __restrict__ A, const float* __restrict__ W,
                        const float* __restrict__ bias, float* __restrict__ out,
                        const float* __restrict__ resid, int M, int K, int N, int dogelu) {
    int t = blockIdx.x * 256 + threadIdx.x;
    if (t >= M * N) return;
    int m = t / N, n = t % N;
    float s = bias ? bias[n] : 0.f;
    const float* ar = A + (size_t)m * K;
    for (int k = 0; k < K; ++k) s += ar[k] * W[(size_t)k * N + n];
    if (dogelu) s = gelu_exact(s);
    if (resid) s += resid[t];
    out[t] = s;
}

__global__ __launch_bounds__(1024) void k_ball_attn(const float* __restrict__ qkv,
                                                    float* __restrict__ out,
                                                    int L, int c, int heads) {
    int ball = blockIdx.x, h = blockIdx.y;
    int hd = c / heads;
    __shared__ float qs[512], ks[512], vs[512], sc[1024];
    int t = threadIdx.x, nt = blockDim.x;     // nt == L*L
    for (int idx = t; idx < L * hd; idx += nt) {
        int l = idx / hd, d = idx % hd;
        const float* base = qkv + ((size_t)(ball * L + l)) * 3 * c + h * hd + d;
        qs[idx] = base[0]; ks[idx] = base[c]; vs[idx] = base[2 * c];
    }
    __syncthreads();
    float scale = rsqrtf((float)hd);
    {
        int l = t / L, m = t % L;
        float s = 0.f;
        for (int d = 0; d < hd; ++d) s += qs[l * hd + d] * ks[m * hd + d];
        sc[l * L + m] = s * scale;
    }
    __syncthreads();
    if (t < L) {
        float mx = -1e30f;
        for (int m = 0; m < L; ++m) mx = fmaxf(mx, sc[t * L + m]);
        float sum = 0.f;
        for (int m = 0; m < L; ++m) { float e = expf(sc[t * L + m] - mx); sc[t * L + m] = e; sum += e; }
        float inv = 1.f / sum;
        for (int m = 0; m < L; ++m) sc[t * L + m] *= inv;
    }
    __syncthreads();
    for (int idx = t; idx < L * hd; idx += nt) {
        int l = idx / hd, d = idx % hd;
        float o = 0.f;
        for (int m = 0; m < L; ++m) o += sc[l * L + m] * vs[m * hd + d];
        out[((size_t)(ball * L + l)) * c + h * hd + d] = o;
    }
}

// ---------------------------------------------------------------------------
// out einsum: out[(b,n),h*64+c] = sum_g sw^T[bh,g,n] * proc[bh,g,c]  (WMMA)
// A = sw^T (f32, converted in-register), B = procT bf16 [bh][c][g]
// ---------------------------------------------------------------------------
__global__ __launch_bounds__(256) void k_out_einsum(const float* __restrict__ swT,
                                                    const __bf16* __restrict__ procT,
                                                    __bf16* __restrict__ out, int Ntok) {
    int bh = blockIdx.y;
    int lane = threadIdx.x & 31, wave = threadIdx.x >> 5;
    int wm = wave & 3, wn = wave >> 2;
    int mB = blockIdx.x * 64 + wm * 16;     // token tile
    int nB = wn * 32;                        // c tile (64 cols total)
    int half = lane >> 4, l16 = lane & 15;
    int am = mB + l16;
    const float*  swb = swT   + (size_t)bh * 64 * Ntok;
    const __bf16* bt0 = procT + (size_t)bh * 4096 + (size_t)(nB + l16) * 64;
    const __bf16* bt1 = bt0 + 16 * 64;
    v8f acc0 = {}; v8f acc1 = {};
    for (int kk = 0; kk < 64; kk += 32) {
        v8u au, bu0, bu1;
#pragma unroll
        for (int v = 0; v < 4; ++v) {
            int k0 = kk + half * 8 + 2 * v;
            int k1 = k0 + 16;
            union { __bf16 hh[2]; unsigned u; } c0, c1;
            c0.hh[0] = (__bf16)swb[(size_t)k0 * Ntok + am];
            c0.hh[1] = (__bf16)swb[(size_t)(k0 + 1) * Ntok + am];
            c1.hh[0] = (__bf16)swb[(size_t)k1 * Ntok + am];
            c1.hh[1] = (__bf16)swb[(size_t)(k1 + 1) * Ntok + am];
            au[v] = c0.u; au[v + 4] = c1.u;
        }
        const unsigned* pb0 = (const unsigned*)bt0;
        const unsigned* pb1 = (const unsigned*)bt1;
#pragma unroll
        for (int v = 0; v < 8; ++v) {
            int kb = kk + (v >> 2) * 16 + half * 8 + 2 * (v & 3);
            bu0[v] = pb0[kb >> 1];
            bu1[v] = pb1[kb >> 1];
        }
        v16bf a  = __builtin_bit_cast(v16bf, au);
        v16bf b0 = __builtin_bit_cast(v16bf, bu0);
        v16bf b1 = __builtin_bit_cast(v16bf, bu1);
        acc0 = __builtin_amdgcn_wmma_f32_16x16x32_bf16(false, a, false, b0, (short)0, acc0, false, false);
        acc1 = __builtin_amdgcn_wmma_f32_16x16x32_bf16(false, a, false, b1, (short)0, acc1, false, false);
    }
    int b = bh >> 3, h = bh & 7;
#pragma unroll
    for (int r = 0; r < 8; ++r) {
        int n = mB + half * 8 + r;
        size_t rowoff = ((size_t)b * Ntok + n) * 512 + h * 64;
        out[rowoff + nB + l16]      = (__bf16)acc0[r];
        out[rowoff + nB + l16 + 16] = (__bf16)acc1[r];
    }
}

// ---------------------------------------------------------------------------
// Host-side orchestration
// ---------------------------------------------------------------------------
struct BlkW {
    const float *ln1g, *ln1b, *Wqkv, *bqkv, *Wo, *bo, *ln2g, *ln2b, *W1, *b1, *W2, *b2;
};

static inline int cdiv(int a, int b) { return (a + b - 1) / b; }

static void erwin_stack(hipStream_t s, float* x, const BlkW& w, int c, int Mtok, int L,
                        int heads, float* tmp, float* qkvb, float* atto, float* midb) {
    int nballs = Mtok / L;
    for (int d = 0; d < 4; ++d) {
        k_ln_small<<<Mtok, c, 0, s>>>(x, w.ln1g + d * c, w.ln1b + d * c, tmp, c);
        k_sgemm<<<cdiv(Mtok * 3 * c, 256), 256, 0, s>>>(tmp, w.Wqkv + (size_t)d * c * 3 * c,
                                                        w.bqkv + d * 3 * c, qkvb, nullptr,
                                                        Mtok, c, 3 * c, 0);
        k_ball_attn<<<dim3(nballs, heads), L * L, 0, s>>>(qkvb, atto, L, c, heads);
        k_sgemm<<<cdiv(Mtok * c, 256), 256, 0, s>>>(atto, w.Wo + (size_t)d * c * c,
                                                    w.bo + d * c, x, x, Mtok, c, c, 0);
        k_ln_small<<<Mtok, c, 0, s>>>(x, w.ln2g + d * c, w.ln2b + d * c, tmp, c);
        k_sgemm<<<cdiv(Mtok * 4 * c, 256), 256, 0, s>>>(tmp, w.W1 + (size_t)d * c * 4 * c,
                                                        w.b1 + d * 4 * c, midb, nullptr,
                                                        Mtok, c, 4 * c, 1);
        k_sgemm<<<cdiv(Mtok * c, 256), 256, 0, s>>>(midb, w.W2 + (size_t)d * 4 * c * c,
                                                    w.b2 + d * c, x, x, Mtok, 4 * c, c, 0);
    }
}

extern "C" void kernel_launch(void* const* d_in, const int* in_sizes, int n_in,
                              void* d_out, int out_size, void* d_ws, size_t ws_size,
                              hipStream_t stream) {
    const int B = 2, Ntok = 16384, C = 512, H = 8, DH = 64, G = 64;
    const int BN = B * Ntok;            // 32768 token rows
    const int BHN = B * H * Ntok;       // 262144 (b,h,n) rows
    const int BH = B * H;               // 16 erwin groups
    const int NCH = 16;                 // n-chunks for es reduction

    auto f = [&](int i) { return (const float*)d_in[i]; };
    const float* fx        = f(0);
    const float* ln1_g     = f(1);  const float* ln1_b = f(2);
    const float* ln2_g     = f(3);  const float* ln2_b = f(4);
    const float* Wx        = f(5);  const float* bx    = f(6);
    const float* Wslice    = f(7);  const float* bslice = f(8);
    const float* w_ada     = f(9);  const float* b_ada  = f(10);
    const float* Wout_attn = f(11); const float* bout_attn = f(12);
    const float* mlp_W1    = f(13); const float* mlp_b1 = f(14);
    const float* mlp_W2    = f(15); const float* mlp_b2 = f(16);
    const float* Wemb      = f(17); const float* bemb   = f(18);
    BlkW enc1 = { f(19), f(20), f(21), f(22), f(23), f(24), f(25), f(26), f(27), f(28), f(29), f(30) };
    const float* Wpool     = f(31); const float* bpool  = f(32);
    BlkW enc2 = { f(33), f(34), f(35), f(36), f(37), f(38), f(39), f(40), f(41), f(42), f(43), f(44) };
    const float* Wun       = f(45); const float* bun    = f(46);
    BlkW dec  = { f(47), f(48), f(49), f(50), f(51), f(52), f(53), f(54), f(55), f(56), f(57), f(58) };
    const float* Wout_er   = f(59); const float* bout_er = f(60);

    // ---- workspace carve-out --------------------------------------------
    char* w = (char*)d_ws;
    auto alloc = [&](size_t bytes) { char* p = w; w += (bytes + 255) & ~(size_t)255; return p; };
    __bf16* xln    = (__bf16*)alloc((size_t)BN * C * 2);
    __bf16* WxT    = (__bf16*)alloc((size_t)C * C * 2);
    __bf16* WsT    = (__bf16*)alloc((size_t)DH * G * 2);
    __bf16* WoT    = (__bf16*)alloc((size_t)C * C * 2);
    __bf16* W1T    = (__bf16*)alloc((size_t)C * 4 * C * 2);
    __bf16* W2T    = (__bf16*)alloc((size_t)4 * C * C * 2);
    float*  xp_f   = (float*)alloc((size_t)BN * C * 4);
    __bf16* xp_h   = (__bf16*)alloc((size_t)BN * C * 2);
    float*  invt   = (float*)alloc((size_t)BHN * 4);
    float*  swT    = (float*)alloc((size_t)BHN * G * 4);
    float*  es     = (float*)alloc((size_t)BH * G * DH * 4);
    float*  snorm  = (float*)alloc((size_t)BH * G * 4);
    int*    rankb  = (int*)alloc((size_t)BH * G * 4);
    float*  xg     = (float*)alloc((size_t)BH * G * DH * 4);
    float*  xer    = (float*)alloc((size_t)BH * G * DH * 4);
    float*  skip   = (float*)alloc((size_t)BH * G * DH * 4);
    float*  tmpb   = (float*)alloc((size_t)BH * G * DH * 4);
    float*  qkvb   = (float*)alloc((size_t)1024 * 192 * 4);
    float*  attob  = (float*)alloc((size_t)BH * G * DH * 4);
    float*  midb   = (float*)alloc((size_t)1024 * 256 * 4);
    float*  procg  = (float*)alloc((size_t)BH * G * DH * 4);
    __bf16* procT  = (__bf16*)alloc((size_t)BH * G * DH * 2);
    __bf16* out_h  = (__bf16*)alloc((size_t)BN * C * 2);
    float*  fx2    = (float*)alloc((size_t)BN * C * 4);
    __bf16* h_h    = (__bf16*)alloc((size_t)BN * C * 2);
    __bf16* mid_h  = (__bf16*)alloc((size_t)BN * 4 * C * 2);
    (void)ws_size; (void)in_sizes; (void)n_in; (void)out_size;

    // ---- weight prep (f32 -> bf16 transposed) ---------------------------
    k_wT<<<cdiv(C * C, 256), 256, 0, stream>>>(Wx, WxT, C, C);
    k_wT<<<cdiv(DH * G, 256), 256, 0, stream>>>(Wslice, WsT, DH, G);
    k_wT<<<cdiv(C * C, 256), 256, 0, stream>>>(Wout_attn, WoT, C, C);
    k_wT<<<cdiv(C * 4 * C, 256), 256, 0, stream>>>(mlp_W1, W1T, C, 4 * C);
    k_wT<<<cdiv(4 * C * C, 256), 256, 0, stream>>>(mlp_W2, W2T, 4 * C, C);

    // ---- slice attention front-end --------------------------------------
    k_ln_big<<<BN / 8, 256, 0, stream>>>(fx, ln1_g, ln1_b, xln, BN);
    // xp = LN(fx) @ Wx + bx  -> [ (b,n,h), dh ] both f32 & bf16
    k_wmma_gemm<EPI_XP><<<(BN / 128) * (C / 64), 256, 0, stream>>>(
        xln, WxT, bx, xp_f, xp_h, nullptr, BN, C, C, Ntok, H);
    k_temp<<<cdiv(BHN, 8), 256, 0, stream>>>(xp_f, w_ada, b_ada, invt, Ntok, H, BHN);
    // logits/temp -> sw^T [b,h,g,n]
    k_wmma_gemm<EPI_LOGIT_T><<<(BHN / 128) * (G / 64), 256, 0, stream>>>(
        xp_h, WsT, bslice, swT, nullptr, invt, BHN, DH, G, Ntok, H);
    k_softmax<<<BH * G, 256, 0, stream>>>(swT, Ntok);
    k_zero<<<cdiv(BH * G * DH, 256), 256, 0, stream>>>(es, BH * G * DH);
    k_zero<<<cdiv(BH * G, 256), 256, 0, stream>>>(snorm, BH * G);
    k_es_partial<<<dim3(4, BH, NCH), 256, 0, stream>>>(xp_f, swT, es, snorm, Ntok, H, NCH);
    k_es_div<<<cdiv(BH * G * DH, 256), 256, 0, stream>>>(es, snorm);

    // ---- erwin hierarchical stack (tiny) --------------------------------
    k_rank<<<BH, 64, 0, stream>>>(es, rankb);
    k_gather<<<cdiv(BH * G * DH, 256), 256, 0, stream>>>(es, rankb, xg);
    k_sgemm<<<cdiv(1024 * 64, 256), 256, 0, stream>>>(xg, Wemb, bemb, xer, nullptr, 1024, 64, 64, 0);
    erwin_stack(stream, xer, enc1, 64, 1024, 32, H, tmpb, qkvb, attob, midb);
    (void)hipMemcpyAsync(skip, xer, (size_t)1024 * 64 * 4, hipMemcpyDeviceToDevice, stream);
    k_sgemm<<<cdiv(512 * 128, 256), 256, 0, stream>>>(xer, Wpool, bpool, tmpb, nullptr, 512, 128, 128, 0);
    (void)hipMemcpyAsync(xer, tmpb, (size_t)512 * 128 * 4, hipMemcpyDeviceToDevice, stream);
    erwin_stack(stream, xer, enc2, 128, 512, 16, H, tmpb, qkvb, attob, midb);
    k_sgemm<<<cdiv(512 * 128, 256), 256, 0, stream>>>(xer, Wun, bun, tmpb, skip, 512, 128, 128, 0);
    (void)hipMemcpyAsync(xer, tmpb, (size_t)512 * 128 * 4, hipMemcpyDeviceToDevice, stream);
    erwin_stack(stream, xer, dec, 64, 1024, 32, H, tmpb, qkvb, attob, midb);
    k_sgemm<<<cdiv(1024 * 64, 256), 256, 0, stream>>>(xer, Wout_er, bout_er, procg, nullptr, 1024, 64, 64, 0);
    k_proc_t<<<cdiv(BH * G * DH, 256), 256, 0, stream>>>(procg, rankb, procT);

    // ---- de-slice + output projection + MLP -----------------------------
    k_out_einsum<<<dim3(Ntok / 64, BH), 256, 0, stream>>>(swT, procT, out_h, Ntok);
    k_wmma_gemm<EPI_RESID_F32><<<(BN / 128) * (C / 64), 256, 0, stream>>>(
        out_h, WoT, bout_attn, fx2, nullptr, fx, BN, C, C, Ntok, H);
    k_ln_big<<<BN / 8, 256, 0, stream>>>(fx2, ln2_g, ln2_b, h_h, BN);
    k_wmma_gemm<EPI_GELU_BF16><<<(BN / 128) * (4 * C / 64), 256, 0, stream>>>(
        h_h, W1T, mlp_b1, nullptr, mid_h, nullptr, BN, C, 4 * C, Ntok, H);
    k_wmma_gemm<EPI_RESID_F32><<<(BN / 128) * (C / 64), 256, 0, stream>>>(
        mid_h, W2T, mlp_b2, (float*)d_out, nullptr, fx2, BN, 4 * C, C, Ntok, H);
}
